// MeanAggregator_48275432407147
// MI455X (gfx1250) — compile-verified, matrix operand order checked by reading, not run on hardware
//
#include <hip/hip_runtime.h>
#include <hip/hip_bf16.h>

typedef __attribute__((ext_vector_type(16))) _Float16 v16h;
typedef __attribute__((ext_vector_type(8)))  _Float16 v8h;
typedef __attribute__((ext_vector_type(4)))  _Float16 v4h;
typedef __attribute__((ext_vector_type(8)))  float    v8f;

#define NUM_NODES 100000
#define NBATCH    50000
#define KNEI      7
#define FDIM      128
#define EDIM      256
#define KDIM      896   // 7*128
#define NKSTEP    28    // 896 / 32
#define BN_EPS    1e-5f
#define LRELU     0.01f

union AFrag { v16h v; v8h h[2]; };

// ---------------------------------------------------------------------------
// 1) features f32 -> f16 (gather source becomes 25.6MB, L2-resident)
// ---------------------------------------------------------------------------
__global__ __launch_bounds__(256) void sage_cvt_feat(const float* __restrict__ in,
                                                     _Float16* __restrict__ out) {
    size_t i = ((size_t)blockIdx.x * 256 + threadIdx.x) * 4;   // 12.8M elems total
    float4 v = *(const float4*)(in + i);
    v4h h = { (_Float16)v.x, (_Float16)v.y, (_Float16)v.z, (_Float16)v.w };
    *(v4h*)(out + i) = h;
}

// ---------------------------------------------------------------------------
// 2) weight [896][256] f32 -> Wt [256][896] f16 (transposed for B fragments)
// ---------------------------------------------------------------------------
__global__ __launch_bounds__(256) void sage_cvt_wt(const float* __restrict__ w,
                                                   _Float16* __restrict__ wt) {
    int n = blockIdx.x;                       // output column 0..255
    #pragma unroll
    for (int it = 0; it < 4; ++it) {
        int k = it * 256 + threadIdx.x;
        if (k < KDIM)
            wt[(size_t)n * KDIM + k] = (_Float16)w[(size_t)k * EDIM + n];
    }
}

// ---------------------------------------------------------------------------
// 3) fused gather + GEMM: x[50000][256] = concat(feat[neigh]) @ W
//    block = 8 waves; tile = 32 rows x 256 cols; wave holds 4 accumulators.
//    Fully unrolled 28 K-steps, software-pipelined one step ahead so the
//    next step's A + 4 B fragments load while the current 4 WMMAs execute.
// ---------------------------------------------------------------------------
__global__ __launch_bounds__(256) void sage_gemm_wmma(
    const _Float16* __restrict__ feat,   // [NUM_NODES][128] f16
    const _Float16* __restrict__ wt,     // [256][896] f16 (W^T)
    const int*      __restrict__ neighs, // [NBATCH][7]
    float*          __restrict__ out)    // [NBATCH][256] f32 (pre-BN)
{
    const int lane    = threadIdx.x & 31;
    const int wave    = threadIdx.x >> 5;
    const int rowTile = wave & 1;        // 2 row tiles of 16
    const int colTile = wave >> 1;       // 4 col tiles of 16 (x4 groups of 64)
    const int lh      = lane & 15;
    const int kb      = (lane < 16) ? 0 : 8;   // ISA 16-bit A/B lane K-base

    const int rowBase = blockIdx.x * 32 + rowTile * 16;
    int arow = rowBase + lh;
    if (arow >= NBATCH) arow = NBATCH - 1;     // clamp: keep EXEC all-1s for WMMA

    int nidx[KNEI];
    #pragma unroll
    for (int j = 0; j < KNEI; ++j)
        nidx[j] = neighs[(size_t)arow * KNEI + j];

    // warm the 7 gathered feature rows (temporal prefetch)
    #pragma unroll
    for (int j = 0; j < KNEI; ++j)
        __builtin_prefetch(feat + (size_t)nidx[j] * FDIM, 0, 3);

    // per-neighbor A base pointers (static index under full unroll)
    const _Float16* aps[KNEI];
    #pragma unroll
    for (int j = 0; j < KNEI; ++j)
        aps[j] = feat + (size_t)nidx[j] * FDIM + kb;

    // per-wave B base: column n0 = colTile*16 + lh, K offset kb
    const _Float16* wtb = wt + (size_t)(colTile * 16 + lh) * KDIM + kb;

    v8f acc[4];
    #pragma unroll
    for (int g = 0; g < 4; ++g)
        acc[g] = (v8f){0.f,0.f,0.f,0.f,0.f,0.f,0.f,0.f};

    // ---- prologue: load fragments for step 0 ----
    AFrag aCur, bCur[4];
    {
        const _Float16* ap = aps[0];
        aCur.h[0] = *(const v8h*)ap;
        aCur.h[1] = *(const v8h*)(ap + 16);
        #pragma unroll
        for (int g = 0; g < 4; ++g) {
            const _Float16* bp = wtb + (size_t)g * 64 * KDIM;
            bCur[g].h[0] = *(const v8h*)bp;
            bCur[g].h[1] = *(const v8h*)(bp + 16);
        }
    }

    // ---- main: step s uses (aCur,bCur), step s+1 loads into (aNxt,bNxt) ----
    #pragma unroll
    for (int s = 0; s < NKSTEP; ++s) {
        AFrag aNxt, bNxt[4];
        if (s + 1 < NKSTEP) {
            const int jn = (s + 1) >> 2;           // neighbor of next step
            const int kn = (s + 1) & 3;            // 32-chunk within feature row
            const _Float16* ap = aps[jn] + kn * 32;
            aNxt.h[0] = *(const v8h*)ap;
            aNxt.h[1] = *(const v8h*)(ap + 16);
            #pragma unroll
            for (int g = 0; g < 4; ++g) {
                const _Float16* bp = wtb + (size_t)g * 64 * KDIM + (s + 1) * 32;
                bNxt[g].h[0] = *(const v8h*)bp;
                bNxt[g].h[1] = *(const v8h*)(bp + 16);
            }
        }
        #pragma unroll
        for (int g = 0; g < 4; ++g)
            acc[g] = __builtin_amdgcn_wmma_f32_16x16x32_f16(
                         false, aCur.v, false, bCur[g].v, (short)0, acc[g],
                         false, false);
        if (s + 1 < NKSTEP) {
            aCur = aNxt;
            #pragma unroll
            for (int g = 0; g < 4; ++g) bCur[g] = bNxt[g];
        }
    }

    // ---- epilogue: C/D layout — lane holds col lh, rows r(+8 for hi lanes) ----
    const int srowBase = rowBase + ((lane >> 4) << 3);
    #pragma unroll
    for (int r = 0; r < 8; ++r) {
        const int srow = srowBase + r;
        if (srow < NBATCH) {
            float* o = out + (size_t)srow * EDIM + colTile * 16 + lh;
            o[0]   = acc[0][r];
            o[64]  = acc[1][r];
            o[128] = acc[2][r];
            o[192] = acc[3][r];
        }
    }
}

// ---------------------------------------------------------------------------
// 4) BN stats stage 1: per-block partial column sums (deterministic)
// ---------------------------------------------------------------------------
__global__ __launch_bounds__(256) void sage_bn_stats1(const float* __restrict__ x,
                                                      float* __restrict__ psum,
                                                      float* __restrict__ psumsq) {
    const int t = threadIdx.x;           // column
    const int b = blockIdx.x;            // row chunk (256 chunks of 196 rows)
    const int r0 = b * 196;
    const int r1 = (r0 + 196 < NBATCH) ? r0 + 196 : NBATCH;
    float s = 0.f, s2 = 0.f;
    for (int r = r0; r < r1; ++r) {
        float v = x[(size_t)r * EDIM + t];
        s += v; s2 += v * v;
    }
    psum[b * EDIM + t]   = s;
    psumsq[b * EDIM + t] = s2;
}

// ---------------------------------------------------------------------------
// 5) BN stats stage 2: final reduce (fixed order) -> fused scale/shift
// ---------------------------------------------------------------------------
__global__ __launch_bounds__(256) void sage_bn_stats2(const float* __restrict__ psum,
                                                      const float* __restrict__ psumsq,
                                                      const float* __restrict__ gamma,
                                                      const float* __restrict__ beta,
                                                      float* __restrict__ ab) {
    const int t = threadIdx.x;
    float s = 0.f, s2 = 0.f;
    for (int b = 0; b < 256; ++b) {
        s  += psum[b * EDIM + t];
        s2 += psumsq[b * EDIM + t];
    }
    const float mu   = s / (float)NBATCH;
    const float var  = s2 / (float)NBATCH - mu * mu;   // biased variance
    const float inv  = rsqrtf(var + BN_EPS);
    const float a    = inv * gamma[t];
    ab[t]        = a;
    ab[EDIM + t] = beta[t] - mu * a;
}

// ---------------------------------------------------------------------------
// 6) BN apply + LeakyReLU, in place on d_out, float4 vectorized
// ---------------------------------------------------------------------------
__global__ __launch_bounds__(256) void sage_bn_apply(float* __restrict__ x,
                                                     const float* __restrict__ ab) {
    size_t gid = (size_t)blockIdx.x * 256 + threadIdx.x;
    size_t i = gid * 4;                  // 12.8M floats total
    const int c = (int)(i & (EDIM - 1));
    float4 v  = *(float4*)(x + i);
    float4 av = *(const float4*)(ab + c);
    float4 bv = *(const float4*)(ab + EDIM + c);
    float y0 = v.x * av.x + bv.x;
    float y1 = v.y * av.y + bv.y;
    float y2 = v.z * av.z + bv.z;
    float y3 = v.w * av.w + bv.w;
    v.x = (y0 >= 0.f) ? y0 : LRELU * y0;
    v.y = (y1 >= 0.f) ? y1 : LRELU * y1;
    v.z = (y2 >= 0.f) ? y2 : LRELU * y2;
    v.w = (y3 >= 0.f) ? y3 : LRELU * y3;
    *(float4*)(x + i) = v;
}

// ---------------------------------------------------------------------------
extern "C" void kernel_launch(void* const* d_in, const int* in_sizes, int n_in,
                              void* d_out, int out_size, void* d_ws, size_t ws_size,
                              hipStream_t stream) {
    (void)in_sizes; (void)n_in; (void)out_size; (void)ws_size;
    const float* raw    = (const float*)d_in[0];   // [100000,128]
    const float* weight = (const float*)d_in[1];   // [896,256]
    const float* gamma  = (const float*)d_in[2];   // [256]
    const float* beta   = (const float*)d_in[3];   // [256]
    const int*   neighs = (const int*)d_in[4];     // [50000,7]
    float* out = (float*)d_out;                    // [50000,256]

    char* ws = (char*)d_ws;
    _Float16* featf16 = (_Float16*)ws;                         // 25,600,000 B
    _Float16* wtf16   = (_Float16*)(ws + 25600000);            //    458,752 B
    float*    psum    = (float*)(ws + 25600000 + 458752);      //    262,144 B
    float*    psumsq  = psum + 256 * EDIM;                     //    262,144 B
    float*    ab      = psumsq + 256 * EDIM;                   //      2,048 B

    sage_cvt_feat <<<12500, 256, 0, stream>>>(raw, featf16);   // 12.8M/4 lanes
    sage_cvt_wt   <<<EDIM,  256, 0, stream>>>(weight, wtf16);
    sage_gemm_wmma<<<(NBATCH + 31) / 32, 256, 0, stream>>>(featf16, wtf16, neighs, out);
    sage_bn_stats1<<<256, 256, 0, stream>>>(out, psum, psumsq);
    sage_bn_stats2<<<1,   256, 0, stream>>>(psum, psumsq, gamma, beta, ab);
    sage_bn_apply <<<12500, 256, 0, stream>>>(out, ab);        // 12.8M/4 lanes
}
